// Head_42889543418420
// MI455X (gfx1250) — matrix operand order repaired and checked by reading, not compiled, not run on hardware
//
#include <hip/hip_runtime.h>

typedef _Float16 v16h __attribute__((ext_vector_type(16)));
typedef _Float16 v8h  __attribute__((ext_vector_type(8)));
typedef float    v8f  __attribute__((ext_vector_type(8)));
typedef float    v4f  __attribute__((ext_vector_type(4)));

#define B_   512
#define T_   128
#define C_   384
#define H_   64
#define SCALE_ 0.4082482904638631f   // 6 ** -0.5  (head_count scaling, faithful to ref)

// LDS row strides (elements), padded so 16B-aligned vector loads hold and banks spread
#define WT_S 392   // W^T  [64][392]  (uses 384)
#define QK_S 72    // Q,K  [128][72]  (uses 64)
#define VT_S 136   // V^T  [64][136]  (uses 128)
#define P_S  136   // P    [128][136] (uses 128)

__device__ __forceinline__ v16h joinh(v8h lo, v8h hi) {
  v16h r;
#pragma unroll
  for (int i = 0; i < 8; ++i) { r[i] = lo[i]; r[i + 8] = hi[i]; }
  return r;
}

// A operand (16x32 f16): lane m = row m; per-lane chunks [base, base+8) and [base+16, base+24)
__device__ __forceinline__ v16h load_a_lds(const _Float16* p) {
  v8h lo = *(const v8h*)(p);
  v8h hi = *(const v8h*)(p + 16);
  return joinh(lo, hi);
}

// B operand (32x16 f16): lane n = column n; per-lane chunks [base, base+8) and [base+16, base+24)
__device__ __forceinline__ v16h load_b_lds(const _Float16* p) {
  v8h lo = *(const v8h*)(p);
  v8h hi = *(const v8h*)(p + 16);
  return joinh(lo, hi);
}

// A operand straight from global f32 (x matrix), converted to f16
__device__ __forceinline__ v16h load_a_global_f32(const float* p) {
  v4f f0 = *(const v4f*)(p);
  v4f f1 = *(const v4f*)(p + 4);
  v4f f2 = *(const v4f*)(p + 16);
  v4f f3 = *(const v4f*)(p + 20);
  v16h a;
#pragma unroll
  for (int i = 0; i < 4; ++i) {
    a[i]      = (_Float16)f0[i];
    a[4 + i]  = (_Float16)f1[i];
    a[8 + i]  = (_Float16)f2[i];
    a[12 + i] = (_Float16)f3[i];
  }
  return a;
}

__device__ __forceinline__ v8f wmma16x16x32(v16h a, v16h b, v8f c) {
  // (neg_a, A, neg_b, B, c_mod, C, reuse_a, reuse_b)
  return __builtin_amdgcn_wmma_f32_16x16x32_f16(false, a, false, b, (short)0, c,
                                                false, false);
}

__global__ __launch_bounds__(256) void attn_head_kernel(
    const float* __restrict__ x, const float* __restrict__ Wq,
    const float* __restrict__ Wk, const float* __restrict__ Wv,
    float* __restrict__ out) {
  __shared__ _Float16 sWt[H_ * WT_S];   // 50,176 B  (current W, transposed)
  __shared__ _Float16 sQ [T_ * QK_S];   // 18,432 B
  __shared__ _Float16 sK [T_ * QK_S];   // 18,432 B
  __shared__ _Float16 sVt[H_ * VT_S];   // 17,408 B  (V transposed)
  __shared__ _Float16 sP [T_ * P_S];    // 34,816 B  -> total 139,264 B

  const int b    = blockIdx.x;
  const int tid  = threadIdx.x;
  const int w    = tid >> 5;        // wave id 0..7 -> owns rows 16w..16w+15
  const int lane = tid & 31;
  const int lp   = lane & 15;
  const int hg   = (lane >> 4) & 1; // half-group within wave
  const int rowbase = w * 16;

  const float* xb   = x + (size_t)b * T_ * C_;
  const float* xrow = xb + (size_t)(rowbase + lp) * C_;   // A-operand row for this lane

  const float* Ws[3] = {Wq, Wk, Wv};

  // ---------------- Phase 1: Q, K, V projections (x @ W) ----------------
#pragma unroll
  for (int m = 0; m < 3; ++m) {
    // stage W^T into LDS: Wt[h][c] = W[c][h]  (coalesced global read)
    for (int i = tid; i < C_ * H_; i += 256) {
      int c = i >> 6, h = i & 63;
      sWt[h * WT_S + c] = (_Float16)Ws[m][i];
    }
    __syncthreads();

    v8f acc[4];
#pragma unroll
    for (int n = 0; n < 4; ++n)
#pragma unroll
      for (int i = 0; i < 8; ++i) acc[n][i] = 0.f;

#pragma unroll
    for (int c0 = 0; c0 < C_; c0 += 32) {
      v16h a = load_a_global_f32(xrow + c0 + 8 * hg);
#pragma unroll
      for (int n = 0; n < 4; ++n) {
        v16h bb = load_b_lds(sWt + (n * 16 + lp) * WT_S + c0 + 8 * hg);
        acc[n] = wmma16x16x32(a, bb, acc[n]);
      }
    }

    // write result tiles (C/D layout: lane -> col n*16+lp, vgpr r -> row r+8*hg)
#pragma unroll
    for (int n = 0; n < 4; ++n)
#pragma unroll
      for (int r = 0; r < 8; ++r) {
        int row = rowbase + r + 8 * hg;
        int col = n * 16 + lp;
        _Float16 v = (_Float16)acc[n][r];
        if (m == 0)      sQ[row * QK_S + col] = v;
        else if (m == 1) sK[row * QK_S + col] = v;
        else             sVt[col * VT_S + row] = v;   // store V transposed
      }
    __syncthreads();   // Wt reuse + publish K / Vt to all waves
  }

  // ---------------- Phase 2: S = Q K^T (scaled, causal) ----------------
  v8f s[8];
#pragma unroll
  for (int st = 0; st < 8; ++st)
#pragma unroll
    for (int i = 0; i < 8; ++i) s[st][i] = 0.f;

  const _Float16* qrow = sQ + (rowbase + lp) * QK_S;
#pragma unroll
  for (int c0 = 0; c0 < H_; c0 += 32) {
    v16h a = load_a_lds(qrow + c0 + 8 * hg);
#pragma unroll
    for (int st = 0; st < 8; ++st) {
      v16h bb = load_b_lds(sK + (st * 16 + lp) * QK_S + c0 + 8 * hg);
      s[st] = wmma16x16x32(a, bb, s[st]);
    }
  }

  // ---------------- Phase 3: softmax per row (C/D register layout) ----------------
#pragma unroll
  for (int r = 0; r < 8; ++r) {
    const int rowg = rowbase + r + 8 * hg;
    float mx = -__builtin_inff();
#pragma unroll
    for (int st = 0; st < 8; ++st) {
      int colg = st * 16 + lp;
      float v = s[st][r] * SCALE_;
      v = (colg > rowg) ? -__builtin_inff() : v;   // causal mask
      s[st][r] = v;
      mx = fmaxf(mx, v);
    }
#pragma unroll
    for (int msk = 8; msk >= 1; msk >>= 1)         // reduce within 16-lane group
      mx = fmaxf(mx, __shfl_xor(mx, msk, 32));
    float sum = 0.f;
#pragma unroll
    for (int st = 0; st < 8; ++st) {
      float e = __expf(s[st][r] - mx);             // masked -> exp(-inf) = 0
      s[st][r] = e;
      sum += e;
    }
#pragma unroll
    for (int msk = 8; msk >= 1; msk >>= 1)
      sum += __shfl_xor(sum, msk, 32);
    float inv = 1.f / sum;
#pragma unroll
    for (int st = 0; st < 8; ++st)
      sP[rowg * P_S + st * 16 + lp] = (_Float16)(s[st][r] * inv);
  }
  // sP rows are wave-private (written+read by same wave): no barrier needed.
  // sVt was published by the Phase-1 trailing barrier.

  // ---------------- Phase 4: O = P V ----------------
  v8f o[4];
#pragma unroll
  for (int n = 0; n < 4; ++n)
#pragma unroll
    for (int i = 0; i < 8; ++i) o[n][i] = 0.f;

  const _Float16* prow = sP + (rowbase + lp) * P_S;
#pragma unroll
  for (int s0 = 0; s0 < T_; s0 += 32) {
    v16h a = load_a_lds(prow + s0 + 8 * hg);
#pragma unroll
    for (int n = 0; n < 4; ++n) {
      v16h bb = load_b_lds(sVt + (n * 16 + lp) * VT_S + s0 + 8 * hg);
      o[n] = wmma16x16x32(a, bb, o[n]);
    }
  }

  float* outb = out + (size_t)b * T_ * H_;
#pragma unroll
  for (int n = 0; n < 4; ++n)
#pragma unroll
    for (int r = 0; r < 8; ++r)
      outb[(size_t)(rowbase + r + 8 * hg) * H_ + n * 16 + lp] = o[n][r];
}

extern "C" void kernel_launch(void* const* d_in, const int* in_sizes, int n_in,
                              void* d_out, int out_size, void* d_ws, size_t ws_size,
                              hipStream_t stream) {
  const float* x  = (const float*)d_in[0];
  const float* Wq = (const float*)d_in[1];
  const float* Wk = (const float*)d_in[2];
  const float* Wv = (const float*)d_in[3];
  float* out = (float*)d_out;
  attn_head_kernel<<<dim3(B_), dim3(256), 0, stream>>>(x, Wq, Wk, Wv, out);
}
